// SimpleSNN_30683246363074
// MI455X (gfx1250) — compile-verified, matrix operand order checked by reading, not run on hardware
//
#include <hip/hip_runtime.h>

// ---------------------------------------------------------------------------
// Fused 2-layer leaky SNN for MI455X (gfx1250, wave32, WMMA).
//   T=100, B=512, NI=784, NH=1024, NO=10, beta=0.9, thresh=1.0
// One block = one 16-row batch tile, persistent over all T steps.
// mem1/mem2 state held in VGPRs in WMMA C-layout. bf16 inputs, f32 accum.
// ---------------------------------------------------------------------------

typedef __bf16 bf16;
typedef __attribute__((ext_vector_type(16))) __bf16 v16bf;
typedef __attribute__((ext_vector_type(8)))  __bf16 v8bf;
typedef __attribute__((ext_vector_type(4)))  __bf16 v4bf;
typedef __attribute__((ext_vector_type(8)))  float  v8f;

union Frag16 { v16bf v; v8bf h[2]; };

constexpr int   kT    = 100;
constexpr int   kB    = 512;
constexpr int   kNI   = 784;
constexpr int   kNI4  = kNI / 4;   // 196 float4 per row
constexpr int   kNIP  = 800;       // K padded to 25 * 32
constexpr int   kNH   = 1024;
constexpr int   kNO   = 10;
constexpr float kBeta = 0.9f;
constexpr float kThr  = 1.0f;

constexpr int kBT      = 16;   // batch rows per block (one WMMA M-tile)
constexpr int kTilesPW = 8;    // 8 waves * 8 tiles * 16 = NH

// LDS strides in bf16 halfs; multiples of 8 (16B vector loads) with
// per-row bank rotation (808*2B = 404 dw -> 20 banks/row; 1048*2B -> 12).
constexpr int kAStride = 808;
constexpr int kSStride = 1048;

// --------------------------- prep: fp32 -> bf16 ----------------------------
// W1b: [NH][NIP] bf16, K-padded with zeros.  (B-fragment reads contiguous K.)
// W2b: [16][NH]  bf16, N-padded rows 10..15 = 0.
__global__ __launch_bounds__(256) void snn_prep(const float* __restrict__ W1,
                                                const float* __restrict__ W2,
                                                bf16* __restrict__ W1b,
                                                bf16* __restrict__ W2b) {
  int idx = blockIdx.x * 256 + threadIdx.x;
  const int n1 = kNH * kNIP;
  if (idx < n1) {
    int n = idx / kNIP, k = idx - n * kNIP;
    float v = (k < kNI) ? W1[n * kNI + k] : 0.f;
    W1b[idx] = (bf16)v;
  } else {
    int j = idx - n1;
    if (j < 16 * kNH) {
      int n = j / kNH, k = j - n * kNH;
      float v = (n < kNO) ? W2[n * kNH + k] : 0.f;
      W2b[j] = (bf16)v;
    }
  }
}

// ------------------------------ fused SNN ----------------------------------
__global__ __launch_bounds__(256) void snn_fused(const float* __restrict__ x,
                                                 const float* __restrict__ b1,
                                                 const float* __restrict__ b2,
                                                 const bf16* __restrict__ W1b,
                                                 const bf16* __restrict__ W2b,
                                                 float* __restrict__ out) {
  __shared__ __align__(16) bf16 sA[kBT * kAStride];  // x_t tile, bf16
  __shared__ __align__(16) bf16 sS[kBT * kSStride];  // spk1 tile, bf16

  const int tid   = threadIdx.x;
  const int lane  = tid & 31;
  const int wave  = tid >> 5;
  const int b0    = blockIdx.x * kBT;        // first batch row of this block
  const int lmod  = lane & 15;               // N index inside a 16-wide tile
  const int lhalf = (lane < 16) ? 0 : 8;     // K-chunk select / C-row offset

  // Zero the K-pad columns of sA once (784..799 for each of 16 rows).
  {
    int m = tid >> 4, k = kNI + (tid & 15);
    sA[m * kAStride + k] = (bf16)0.f;
  }

  // Per-lane biases (loop-invariant).
  float bias1[kTilesPW];
#pragma unroll
  for (int j = 0; j < kTilesPW; ++j)
    bias1[j] = b1[(wave * kTilesPW + j) * 16 + lmod];
  const float bias2 = (lmod < kNO) ? b2[lmod] : 0.f;

  const v8f vzero = {0.f, 0.f, 0.f, 0.f, 0.f, 0.f, 0.f, 0.f};

  // Persistent membrane state in registers (WMMA C-layout).
  v8f mem1[kTilesPW];
#pragma unroll
  for (int j = 0; j < kTilesPW; ++j) mem1[j] = vzero;
  v8f mem2 = vzero;   // only wave 0's copy is meaningful

#pragma unroll 1
  for (int t = 0; t < kT; ++t) {
    // ---- stage x[t, b0:b0+16, :] into LDS as bf16, vectorized 4-wide ------
    // global: float4 (16B) loads; LDS: packed 4x bf16 (8B) stores.
    const float* xt = x + (size_t)t * kB * kNI + (size_t)b0 * kNI;
    const float4* xt4 = (const float4*)xt;
    for (int i4 = tid; i4 < kBT * kNI4; i4 += 256) {
      int m  = i4 / kNI4;
      int k4 = i4 - m * kNI4;
      float4 v = xt4[i4];
      v4bf o = {(bf16)v.x, (bf16)v.y, (bf16)v.z, (bf16)v.w};
      *(v4bf*)&sA[m * kAStride + 4 * k4] = o;
    }
    // Prefetch next step's x tile into cache while this step computes.
    if (t + 1 < kT) {
      const char* nx =
          (const char*)(x + (size_t)(t + 1) * kB * kNI + (size_t)b0 * kNI);
      // 16 rows * 3136 B = 50176 B; one 128B line per thread per iteration.
      for (int off = tid * 128; off < kBT * kNI * 4; off += 256 * 128)
        __builtin_prefetch(nx + off, 0, 0);
    }
    __syncthreads();

    // ---- layer 1: cur1 = x_t @ W1^T   (25 k-steps of 32, 8 N-tiles/wave) --
    v8f acc[kTilesPW];
#pragma unroll
    for (int j = 0; j < kTilesPW; ++j) acc[j] = vzero;

#pragma unroll 1
    for (int kk = 0; kk < kNIP / 32; ++kk) {
      const int k0 = kk * 32 + lhalf;
      Frag16 a;   // A: lane holds row M=lmod, halfs {k0..k0+7, k0+16..k0+23}
      a.h[0] = *(const v8bf*)&sA[lmod * kAStride + k0];
      a.h[1] = *(const v8bf*)&sA[lmod * kAStride + k0 + 16];
#pragma unroll
      for (int j = 0; j < kTilesPW; ++j) {
        const int n = (wave * kTilesPW + j) * 16 + lmod;
        const bf16* wp = W1b + (size_t)n * kNIP + k0;   // L2-resident
        Frag16 bm;  // B: lane holds W1 row n (= K-major), same chunking
        bm.h[0] = *(const v8bf*)wp;
        bm.h[1] = *(const v8bf*)(wp + 16);
        acc[j] = __builtin_amdgcn_wmma_f32_16x16x32_bf16(
            false, a.v, false, bm.v, (short)0, acc[j], false, false);
      }
    }

    // ---- membrane 1 update + spike; spk1 -> LDS (bf16, exact 0/1) --------
#pragma unroll
    for (int j = 0; j < kTilesPW; ++j) {
      const int n = (wave * kTilesPW + j) * 16 + lmod;
#pragma unroll
      for (int r = 0; r < 8; ++r) {
        const float old = mem1[j][r];
        const float nm  = kBeta * old + acc[j][r] + bias1[j]
                          - ((old > kThr) ? kThr : 0.f);
        mem1[j][r] = nm;
        const int m = r + lhalf;                 // C-layout row
        sS[m * kSStride + n] = (bf16)((nm > kThr) ? 1.f : 0.f);
      }
    }
    __syncthreads();

    // ---- layer 2 (wave 0): cur2 = spk1 @ W2p^T, one 16x16 tile, K=1024 ---
    // Other waves overlap this with staging of x[t+1]; boost wave 0's issue
    // priority since its serial WMMA chain is the per-step critical path.
    if (wave == 0) {
      __builtin_amdgcn_s_setprio(3);
      v8f c2 = vzero;
#pragma unroll 1
      for (int kk = 0; kk < kNH / 32; ++kk) {
        const int k0 = kk * 32 + lhalf;
        Frag16 a2, w2f;
        a2.h[0] = *(const v8bf*)&sS[lmod * kSStride + k0];
        a2.h[1] = *(const v8bf*)&sS[lmod * kSStride + k0 + 16];
        const bf16* wp = W2b + (size_t)lmod * kNH + k0;
        w2f.h[0] = *(const v8bf*)wp;
        w2f.h[1] = *(const v8bf*)(wp + 16);
        c2 = __builtin_amdgcn_wmma_f32_16x16x32_bf16(
            false, a2.v, false, w2f.v, (short)0, c2, false, false);
      }
#pragma unroll
      for (int r = 0; r < 8; ++r) {
        const float old = mem2[r];
        const float nm  = kBeta * old + c2[r] + bias2
                          - ((old > kThr) ? kThr : 0.f);
        mem2[r] = nm;
        const int m = r + lhalf;
        if (lmod < kNO) {
          // spk2_rec[t, b0+m, lmod]
          out[(size_t)t * kB * kNO + (size_t)(b0 + m) * kNO + lmod] =
              (nm > kThr) ? 1.f : 0.f;
          if (t == kT - 1)   // trailing mem2 output
            out[(size_t)kT * kB * kNO + (size_t)(b0 + m) * kNO + lmod] = nm;
        }
      }
      __builtin_amdgcn_s_setprio(0);
    }
    // No barrier needed here: next staging touches sA only; wave0's sS reads
    // complete before it signals the top-of-loop barrier.
  }
}

// ------------------------------- launcher ----------------------------------
extern "C" void kernel_launch(void* const* d_in, const int* in_sizes, int n_in,
                              void* d_out, int out_size, void* d_ws,
                              size_t ws_size, hipStream_t stream) {
  (void)in_sizes; (void)n_in; (void)out_size; (void)ws_size;
  const float* x  = (const float*)d_in[0];
  const float* W1 = (const float*)d_in[1];
  const float* b1 = (const float*)d_in[2];
  const float* W2 = (const float*)d_in[3];
  const float* b2 = (const float*)d_in[4];
  float* out = (float*)d_out;

  bf16* W1b = (bf16*)d_ws;                     // [1024][800] bf16 = 1.6 MB
  bf16* W2b = W1b + (size_t)kNH * kNIP;        // [16][1024]  bf16 = 32 KB

  const int prepN = kNH * kNIP + 16 * kNH;
  snn_prep<<<(prepN + 255) / 256, 256, 0, stream>>>(W1, W2, W1b, W2b);
  snn_fused<<<kB / kBT, 256, 0, stream>>>(x, b1, b2, W1b, W2b, out);
}
